// SparseMoE_77893526880326
// MI455X (gfx1250) — matrix operand order repaired and checked by reading, not compiled, not run on hardware
//
#include <hip/hip_runtime.h>
#include <hip/hip_bf16.h>
#include <stdint.h>

// Problem constants (fixed by the reference: B=4, S=2048, D=1024, E=8, H=4096, TOP_K=2)
#define T_TOK 8192
#define DMODEL 1024
#define NEXP 8
#define HDIM 4096
#define MTILE 32   // routed tokens per block

typedef __attribute__((ext_vector_type(16))) __bf16 v16bf;
typedef __attribute__((ext_vector_type(16))) unsigned short v16u;
typedef __attribute__((ext_vector_type(8)))  float v8f;

// ---- helpers -------------------------------------------------------------

// round-to-nearest-even f32 -> bf16 (as raw u16)
__device__ __forceinline__ unsigned short f2bf(float f) {
  unsigned u = __builtin_bit_cast(unsigned, f);
  unsigned r = u + 0x7FFFu + ((u >> 16) & 1u);
  return (unsigned short)(r >> 16);
}

union FragU { v16u u; uint4 q[2]; };

// A fragment (16x32 bf16, M x K), row-major source with element stride `stride`.
// ISA layout: lanes 0-15 -> row M=lane, K={k0..k0+7, k0+16..k0+23};
//             lanes 16-31 -> row M=lane-16, K={k0+8..k0+15, k0+24..k0+31}.
__device__ __forceinline__ v16bf load_a_lds(const unsigned short* base, int stride,
                                            int lane, int k0) {
  int row = lane & 15;
  int kh  = (lane & 16) ? 8 : 0;
  const unsigned short* p = base + row * stride + k0 + kh;
  FragU f;
  f.q[0] = *(const uint4*)(p);       // 8 bf16
  f.q[1] = *(const uint4*)(p + 16);  // 8 bf16 at K+16
  return __builtin_bit_cast(v16bf, f.u);
}

// B fragment (32x16 bf16, K x N) from pre-transposed weights stored [N][K]
// lanes 0-15 -> col N=n0+lane, K={k0..k0+7, k0+16..k0+23};
// lanes 16-31 -> col N=n0+lane-16, K={k0+8..k0+15, k0+24..k0+31}.
__device__ __forceinline__ v16bf load_b_glb(const unsigned short* __restrict__ bt,
                                            int ldk, int lane, int k0, int n0) {
  int col = lane & 15;
  int ko  = (lane & 16) ? 8 : 0;
  const unsigned short* p = bt + (size_t)(n0 + col) * ldk + (size_t)(k0 + ko);
  FragU f;
  f.q[0] = *(const uint4*)(p);
  f.q[1] = *(const uint4*)(p + 16);
  return __builtin_bit_cast(v16bf, f.u);
}

// ---- kernel 1: one-time weight transpose + f32->bf16 ---------------------
// w1t layout: [e][h][d]  (B^T for GEMM1: N=h, K=d contiguous)
// w2t layout: [e][d][h]  (B^T for GEMM2: N=d, K=h contiguous)
__global__ __launch_bounds__(256) void convert_weights_kernel(
    const float* __restrict__ W1, const float* __restrict__ W2,
    unsigned short* __restrict__ w1t, unsigned short* __restrict__ w2t) {
  size_t idx = (size_t)blockIdx.x * 256 + threadIdx.x;
  const size_t N = (size_t)NEXP * DMODEL * HDIM;
  if (idx >= N) return;
  {  // w1t[idx] with idx = ((e*H)+h)*D + d  <- W1[((e*D)+d)*H + h]
    size_t e = idx / ((size_t)HDIM * DMODEL);
    size_t r = idx % ((size_t)HDIM * DMODEL);
    size_t h = r / DMODEL, d = r % DMODEL;
    w1t[idx] = f2bf(W1[((e * DMODEL) + d) * HDIM + h]);
  }
  {  // w2t[idx] with idx = ((e*D)+d)*H + h  <- W2[((e*H)+h)*D + d]
    size_t e = idx / ((size_t)DMODEL * HDIM);
    size_t r = idx % ((size_t)DMODEL * HDIM);
    size_t d = r / HDIM, h = r % HDIM;
    w2t[idx] = f2bf(W2[((e * HDIM) + h) * DMODEL + d]);
  }
}

__global__ __launch_bounds__(256) void convert_x_kernel(
    const float* __restrict__ x, unsigned short* __restrict__ xb) {
  size_t idx = (size_t)blockIdx.x * 256 + threadIdx.x;
  if (idx >= (size_t)T_TOK * DMODEL) return;
  xb[idx] = f2bf(x[idx]);
}

// ---- kernel 2: gating (logits -> softmax -> top2 -> expert lists) --------
__global__ __launch_bounds__(256) void gate_kernel(
    const float* __restrict__ x, const float* __restrict__ wg,
    float* __restrict__ cw, int* __restrict__ counts, int* __restrict__ tlist) {
  int t = blockIdx.x * blockDim.x + threadIdx.x;
  if (t >= T_TOK) return;
  float acc[NEXP];
#pragma unroll
  for (int e = 0; e < NEXP; ++e) acc[e] = 0.f;
  const float* xr = x + (size_t)t * DMODEL;
  for (int d = 0; d < DMODEL; ++d) {
    float xv = xr[d];
#pragma unroll
    for (int e = 0; e < NEXP; ++e) acc[e] += xv * wg[d * NEXP + e];
  }
  float m = acc[0];
#pragma unroll
  for (int e = 1; e < NEXP; ++e) m = fmaxf(m, acc[e]);
  float s = 0.f;
#pragma unroll
  for (int e = 0; e < NEXP; ++e) { acc[e] = __expf(acc[e] - m); s += acc[e]; }
  float inv = 1.f / s;
#pragma unroll
  for (int e = 0; e < NEXP; ++e) acc[e] *= inv;
  // top-2 (ties -> lowest index, matching jax top_k)
  int i1 = 0; float p1 = acc[0];
#pragma unroll
  for (int e = 1; e < NEXP; ++e) if (acc[e] > p1) { p1 = acc[e]; i1 = e; }
  int i2 = -1; float p2 = -1.f;
#pragma unroll
  for (int e = 0; e < NEXP; ++e)
    if (e != i1 && acc[e] > p2) { p2 = acc[e]; i2 = e; }
#pragma unroll
  for (int e = 0; e < NEXP; ++e)
    cw[(size_t)t * NEXP + e] = (e == i1) ? p1 : (e == i2) ? p2 : 0.f;
  int s1 = atomicAdd(&counts[i1], 1); tlist[(size_t)i1 * T_TOK + s1] = t;
  int s2 = atomicAdd(&counts[i2], 1); tlist[(size_t)i2 * T_TOK + s2] = t;
}

// ---- kernel 3: fused expert FFN ------------------------------------------
// block = 32 routed tokens of one expert, 256 threads = 8 waves.
// Wave w owns output columns [w*128, w*128+128) for both 16-row halves:
// 16 accumulator tiles (2 row-tiles x 8 col-tiles); every B fragment feeds
// two WMMAs (row-tile reuse) to double FLOP per L2 byte.
#define XSTRIDE (DMODEL + 8)  // bf16 elems; pad to break LDS bank alignment
#define HSTRIDE (128 + 8)

__global__ __launch_bounds__(256) void moe_ffn_kernel(
    const unsigned short* __restrict__ xb,
    const unsigned short* __restrict__ w1t,
    const unsigned short* __restrict__ w2t,
    const float* __restrict__ b1, const float* __restrict__ b2,
    const float* __restrict__ cw, const int* __restrict__ counts,
    const int* __restrict__ tlist, float* __restrict__ out) {
  const int e = blockIdx.y;
  const int tile = blockIdx.x;
  const int cnt = counts[e];
  if (tile * MTILE >= cnt) return;

  __shared__ unsigned short x_lds[MTILE * XSTRIDE];
  __shared__ unsigned short h_lds[MTILE * HSTRIDE];
  __shared__ int   tok_s[MTILE];
  __shared__ float cw_s[MTILE];

  const int lane = threadIdx.x & 31;
  const int wave = threadIdx.x >> 5;

  if (threadIdx.x < MTILE) {
    int slot = tile * MTILE + threadIdx.x;
    int tv = (slot < cnt) ? tlist[(size_t)e * T_TOK + slot]
                          : tlist[(size_t)e * T_TOK + tile * MTILE];
    tok_s[threadIdx.x] = tv;
    cw_s[threadIdx.x]  = (slot < cnt) ? cw[(size_t)tv * NEXP + e] : 0.f;
  }
  __syncthreads();

  // stage 32 gathered x rows (bf16) into LDS: 256 threads x 128 elems
  {
    int row = threadIdx.x >> 3, seg = threadIdx.x & 7;
    const uint4* src = (const uint4*)(xb + (size_t)tok_s[row] * DMODEL + seg * 128);
    uint4* dst = (uint4*)(&x_lds[row * XSTRIDE + seg * 128]);
#pragma unroll
    for (int i = 0; i < 16; ++i) dst[i] = src[i];
  }
  __syncthreads();

  const unsigned short* w1e = w1t + (size_t)e * HDIM * DMODEL;
  const unsigned short* w2e = w2t + (size_t)e * DMODEL * HDIM;

  v8f accy[16];  // [0..7]=rows 0-15 (8 col tiles), [8..15]=rows 16-31
#pragma unroll
  for (int t = 0; t < 16; ++t) accy[t] = {};

  const int col   = lane & 15;
  const int rbase = (lane & 16) ? 8 : 0;

  for (int hc = 0; hc < HDIM; hc += 128) {
    // ---- GEMM1: two 16x16 h tiles per wave = x_tile @ W1[:, hc+wave*16 ..] ----
    v8f acch0 = {}, acch1 = {};
    const int n0h = hc + wave * 16;
#pragma unroll
    for (int k = 0; k < DMODEL; k += 32) {
      v16bf b  = load_b_glb(w1e, DMODEL, lane, k, n0h);
      v16bf a0 = load_a_lds(x_lds, XSTRIDE, lane, k);
      v16bf a1 = load_a_lds(x_lds + 16 * XSTRIDE, XSTRIDE, lane, k);
      acch0 = __builtin_amdgcn_wmma_f32_16x16x32_bf16(
          false, a0, false, b, (short)0, acch0, false, false);
      acch1 = __builtin_amdgcn_wmma_f32_16x16x32_bf16(
          false, a1, false, b, (short)0, acch1, false, false);
    }
    // exact GELU, convert to bf16, publish into shared h slab
    {
      float bias = b1[(size_t)e * HDIM + n0h + col];
#pragma unroll
      for (int i = 0; i < 8; ++i) {
        float v0 = acch0[i] + bias;
        float g0 = 0.5f * v0 * (1.0f + erff(v0 * 0.70710678118f));
        h_lds[(rbase + i) * HSTRIDE + wave * 16 + col] = f2bf(g0);
        float v1 = acch1[i] + bias;
        float g1 = 0.5f * v1 * (1.0f + erff(v1 * 0.70710678118f));
        h_lds[(16 + rbase + i) * HSTRIDE + wave * 16 + col] = f2bf(g1);
      }
    }
    __syncthreads();
    // ---- GEMM2: accumulate y columns of this wave from the h slab ----
#pragma unroll
    for (int kk = 0; kk < 128; kk += 32) {
      v16bf a0 = load_a_lds(h_lds, HSTRIDE, lane, kk);
      v16bf a1 = load_a_lds(h_lds + 16 * HSTRIDE, HSTRIDE, lane, kk);
#pragma unroll
      for (int t = 0; t < 8; ++t) {
        v16bf b = load_b_glb(w2e, HDIM, lane, hc + kk, wave * 128 + t * 16);
        accy[t] = __builtin_amdgcn_wmma_f32_16x16x32_bf16(
            false, a0, false, b, (short)0, accy[t], false, false);
        accy[8 + t] = __builtin_amdgcn_wmma_f32_16x16x32_bf16(
            false, a1, false, b, (short)0, accy[8 + t], false, false);
      }
    }
    __syncthreads();
  }

  // ---- scatter: out[token] += (y + b2) * cw  (tokens hit by <=2 experts) ----
#pragma unroll
  for (int t = 0; t < 8; ++t) {
    const int n0y = wave * 128 + t * 16;
    float bias2 = b2[(size_t)e * DMODEL + n0y + col];
#pragma unroll
    for (int i = 0; i < 8; ++i) {
      int row0 = rbase + i;
      int row1 = 16 + rbase + i;
      float v0 = (accy[t][i] + bias2) * cw_s[row0];
      float v1 = (accy[8 + t][i] + bias2) * cw_s[row1];
      unsafeAtomicAdd(&out[(size_t)tok_s[row0] * DMODEL + n0y + col], v0);
      unsafeAtomicAdd(&out[(size_t)tok_s[row1] * DMODEL + n0y + col], v1);
    }
  }
}

// ---- host launch ---------------------------------------------------------
extern "C" void kernel_launch(void* const* d_in, const int* in_sizes, int n_in,
                              void* d_out, int out_size, void* d_ws, size_t ws_size,
                              hipStream_t stream) {
  const float* x  = (const float*)d_in[0];
  const float* Wg = (const float*)d_in[1];
  const float* W1 = (const float*)d_in[2];
  const float* b1 = (const float*)d_in[3];
  const float* W2 = (const float*)d_in[4];
  const float* b2 = (const float*)d_in[5];
  float* out = (float*)d_out;

  char* ws = (char*)d_ws;
  size_t off = 0;
  auto take = [&](size_t bytes) { size_t o = off; off = (off + bytes + 255) & ~(size_t)255; return o; };
  size_t off_cw  = take((size_t)T_TOK * NEXP * 4);
  size_t off_cnt = take((size_t)NEXP * 4);
  size_t off_tl  = take((size_t)NEXP * T_TOK * 4);
  size_t off_xb  = take((size_t)T_TOK * DMODEL * 2);
  size_t off_w1  = take((size_t)NEXP * DMODEL * HDIM * 2);
  size_t off_w2  = take((size_t)NEXP * DMODEL * HDIM * 2);
  if (off > ws_size) return;  // workspace too small for bf16 staging

  float* cw            = (float*)(ws + off_cw);
  int*   counts        = (int*)(ws + off_cnt);
  int*   tlist         = (int*)(ws + off_tl);
  unsigned short* xb   = (unsigned short*)(ws + off_xb);
  unsigned short* w1t  = (unsigned short*)(ws + off_w1);
  unsigned short* w2t  = (unsigned short*)(ws + off_w2);

  hipMemsetAsync(counts, 0, NEXP * sizeof(int), stream);
  hipMemsetAsync(out, 0, (size_t)T_TOK * DMODEL * sizeof(float), stream);

  {
    size_t n = (size_t)NEXP * DMODEL * HDIM;
    convert_weights_kernel<<<dim3((unsigned)((n + 255) / 256)), 256, 0, stream>>>(W1, W2, w1t, w2t);
  }
  {
    size_t n = (size_t)T_TOK * DMODEL;
    convert_x_kernel<<<dim3((unsigned)((n + 255) / 256)), 256, 0, stream>>>(x, xb);
  }
  gate_kernel<<<dim3(T_TOK / 256), 256, 0, stream>>>(x, Wg, cw, counts, tlist);
  moe_ffn_kernel<<<dim3(T_TOK / MTILE, NEXP), 256, 0, stream>>>(
      xb, w1t, w2t, b1, b2, cw, counts, tlist, out);
}
